// SparseAutoencoder_20220706030113
// MI455X (gfx1250) — matrix operand order repaired and checked by reading, not compiled, not run on hardware
//
#include <hip/hip_runtime.h>
#include <hip/hip_bf16.h>
#include <stdint.h>

// ---------------------------------------------------------------------------
// Sparse autoencoder forward for MI455X (gfx1250, wave32, WMMA).
//   pre = (x - b_dec) @ W_enc^T + b_enc        [4096 x 32768]  <- bf16 WMMA GEMM
//   h   = topk64(pre) with relu                 (radix-select per row)
//   xh  = h @ W_dec^T + b_dec                   (sparse gather via W_dec^T copy)
//   loss= mean((xh - x)^2)
// d_out = [x_hat (4096*2048) | h (4096*32768) | loss (1)]
//
// GEMM pipeline: double-buffered LDS fed by GLOBAL_LOAD_ASYNC_TO_LDS_B128
// (ASYNCcnt-tracked DMA, no VGPR staging), s_wait_asynccnt for software
// pipelining; bf16 W_enc copy (128 MB) is L2-resident across all row blocks.
// ---------------------------------------------------------------------------

#define N_ROWS 4096
#define D_IN   2048
#define D_SAE  32768
#define TOPK   64

typedef __bf16 v16bf __attribute__((ext_vector_type(16)));
typedef __bf16 v8bf  __attribute__((ext_vector_type(8)));
typedef float  v8f   __attribute__((ext_vector_type(8)));

__device__ __forceinline__ unsigned short f2bf(float f) {
  // round-to-nearest-even f32 -> bf16
  unsigned u = __float_as_uint(f);
  unsigned r = 0x7fffu + ((u >> 16) & 1u);
  return (unsigned short)((u + r) >> 16);
}

__device__ __forceinline__ unsigned fkey(float f) {
  // order-preserving monotonic map float -> uint (bigger key == bigger float)
  unsigned u = __float_as_uint(f);
  return (u & 0x80000000u) ? ~u : (u | 0x80000000u);
}

// Async DMA: 16 bytes global -> LDS per lane, tracked by ASYNCcnt.
__device__ __forceinline__ void async_ld16(uint32_t lds_off, const unsigned short* g) {
  asm volatile("global_load_async_to_lds_b128 %0, %1, off"
               :: "v"(lds_off), "v"(g)
               : "memory");
}

// --------------------------- prep kernels ----------------------------------

__global__ __launch_bounds__(256) void prep_x_kernel(
    const float* __restrict__ x, const float* __restrict__ b_dec,
    unsigned short* __restrict__ Xc) {
  const size_t total = (size_t)N_ROWS * D_IN;
  for (size_t i = (size_t)blockIdx.x * blockDim.x + threadIdx.x; i < total;
       i += (size_t)gridDim.x * blockDim.x) {
    int d = (int)(i & (D_IN - 1));
    Xc[i] = f2bf(x[i] - b_dec[d]);
  }
}

__global__ __launch_bounds__(256) void prep_we_kernel(
    const float* __restrict__ W_enc, unsigned short* __restrict__ We) {
  const size_t total = (size_t)D_SAE * D_IN;
  for (size_t i = (size_t)blockIdx.x * blockDim.x + threadIdx.x; i < total;
       i += (size_t)gridDim.x * blockDim.x) {
    We[i] = f2bf(W_enc[i]);
  }
}

// W_dec [D_IN][D_SAE] -> WdT [D_SAE][D_IN]  (tiled LDS transpose)
__global__ __launch_bounds__(256) void transpose_wdec_kernel(
    const float* __restrict__ W_dec, float* __restrict__ WdT) {
  __shared__ float tile[32][33];
  const int tx = threadIdx.x, ty = threadIdx.y;          // block (32, 8)
  const int dbase = blockIdx.x * 32;                     // over D_IN
  const int sbase = blockIdx.y * 32;                     // over D_SAE
#pragma unroll
  for (int j = 0; j < 4; ++j) {
    int d = dbase + ty + 8 * j;
    int s = sbase + tx;
    tile[tx][ty + 8 * j] = W_dec[(size_t)d * D_SAE + s];
  }
  __syncthreads();
#pragma unroll
  for (int j = 0; j < 4; ++j) {
    int s = sbase + ty + 8 * j;
    int d = dbase + tx;
    WdT[(size_t)s * D_IN + d] = tile[ty + 8 * j][tx];
  }
}

// --------------------------- encode GEMM -----------------------------------
// C[128x128] per block, 8 waves, each wave: 32x64 = 2x4 WMMA 16x16 tiles.
// bf16 A (rows of x-b_dec) and bf16 B (rows of W_enc == columns of W_enc^T).

#define BM 128
#define BN 128
#define BK 32
#define LDT 40  // shorts per LDS row (32 data + 8 pad -> conflict-free b128 reads)

__global__ __launch_bounds__(256) void encode_gemm_kernel(
    const unsigned short* __restrict__ Xc,   // [N_ROWS][D_IN] bf16
    const unsigned short* __restrict__ We,   // [D_SAE][D_IN] bf16 (L2-resident)
    const float* __restrict__ b_enc,
    float* __restrict__ pre)                 // [N_ROWS][D_SAE] (h region of d_out)
{
  __shared__ unsigned short As[2][BM * LDT];
  __shared__ unsigned short Bs[2][BN * LDT];

  const int tid  = threadIdx.x;
  const int lane = tid & 31;
  const int wave = tid >> 5;
  const int wr = wave >> 1;   // 0..3 (row group of 32)
  const int wc = wave & 1;    // 0..1 (col group of 64)
  const int row0 = blockIdx.y * BM;
  const int col0 = blockIdx.x * BN;

  v8f acc[2][4] = {};

  // staging map: tile = 128 rows x 4 chunks of 16B (8 bf16); 512 chunks / 256 thr
  const int c0 = tid * 2, c1 = tid * 2 + 1;
  const int r0 = c0 >> 2, q0 = (c0 & 3) * 8;
  const int r1 = c1 >> 2, q1 = (c1 & 3) * 8;

  const unsigned short* gA0 = Xc + (size_t)(row0 + r0) * D_IN + q0;
  const unsigned short* gA1 = Xc + (size_t)(row0 + r1) * D_IN + q1;
  const unsigned short* gB0 = We + (size_t)(col0 + r0) * D_IN + q0;
  const unsigned short* gB1 = We + (size_t)(col0 + r1) * D_IN + q1;

  // LDS byte addresses for this lane's 4 chunks, per buffer
  // (low 32 bits of a generic pointer to LDS == LDS byte address)
  uint32_t lA0[2], lA1[2], lB0[2], lB1[2];
#pragma unroll
  for (int b = 0; b < 2; ++b) {
    lA0[b] = (uint32_t)(size_t)&As[b][r0 * LDT + q0];
    lA1[b] = (uint32_t)(size_t)&As[b][r1 * LDT + q1];
    lB0[b] = (uint32_t)(size_t)&Bs[b][r0 * LDT + q0];
    lB1[b] = (uint32_t)(size_t)&Bs[b][r1 * LDT + q1];
  }

  const int kA0  = (lane < 16) ? 0 : 8;   // A frag: K {0..7,16..23} / {8..15,24..31}
  const int kB0  = (lane < 16) ? 0 : 8;   // B frag: K {0..7,16..23} / {8..15,24..31}
  const int mrow = lane & 15;

  // prologue: tile 0 in flight (4 async-load instructions per wave per tile)
  async_ld16(lA0[0], gA0);
  async_ld16(lA1[0], gA1);
  async_ld16(lB0[0], gB0);
  async_ld16(lB1[0], gB1);

  const int NT = D_IN / BK;  // 64 K-tiles
  for (int kt = 0; kt < NT; ++kt) {
    const int buf = kt & 1;
    if (kt + 1 < NT) {
      // issue tile kt+1 into the other buffer, then wait for tile kt:
      // async loads retire in order, so asynccnt<=4 means tile kt has landed.
      const int ke = (kt + 1) * BK;
      async_ld16(lA0[buf ^ 1], gA0 + ke);
      async_ld16(lA1[buf ^ 1], gA1 + ke);
      async_ld16(lB0[buf ^ 1], gB0 + ke);
      async_ld16(lB1[buf ^ 1], gB1 + ke);
      asm volatile("s_wait_asynccnt 0x4" ::: "memory");
    } else {
      asm volatile("s_wait_asynccnt 0x0" ::: "memory");
    }
    __syncthreads();  // tile kt visible to all waves

    union frag { v16bf v; v8bf h[2]; };
    frag a[2], b[4];
#pragma unroll
    for (int mi = 0; mi < 2; ++mi) {
      const unsigned short* ap = &As[buf][(wr * 32 + mi * 16 + mrow) * LDT];
      a[mi].h[0] = *reinterpret_cast<const v8bf*>(ap + kA0);
      a[mi].h[1] = *reinterpret_cast<const v8bf*>(ap + kA0 + 16);
    }
#pragma unroll
    for (int ni = 0; ni < 4; ++ni) {
      const unsigned short* bp = &Bs[buf][(wc * 64 + ni * 16 + mrow) * LDT];
      b[ni].h[0] = *reinterpret_cast<const v8bf*>(bp + kB0);
      b[ni].h[1] = *reinterpret_cast<const v8bf*>(bp + kB0 + 16);
    }
#pragma unroll
    for (int mi = 0; mi < 2; ++mi)
#pragma unroll
      for (int ni = 0; ni < 4; ++ni)
        acc[mi][ni] = __builtin_amdgcn_wmma_f32_16x16x32_bf16(
            false, a[mi].v, false, b[ni].v, (short)0, acc[mi][ni], false, false);

    __syncthreads();  // all reads of buf done before tile kt+2 overwrites it
  }

  // epilogue: + b_enc, scatter per C-layout (VGPR r: lanes0-15 M=r, lanes16-31 M=8+r)
#pragma unroll
  for (int ni = 0; ni < 4; ++ni) {
    int col = col0 + wc * 64 + ni * 16 + (lane & 15);
    float be = b_enc[col];
#pragma unroll
    for (int mi = 0; mi < 2; ++mi) {
      int rbase = row0 + wr * 32 + mi * 16 + ((lane < 16) ? 0 : 8);
#pragma unroll
      for (int r = 0; r < 8; ++r)
        pre[(size_t)(rbase + r) * D_SAE + col] = acc[mi][ni][r] + be;
    }
  }
}

// --------------------------- top-k (radix select) --------------------------
// One block per row. 4-pass byte histogram finds the 64th-largest key, then a
// deterministic ordered compaction writes (idx, relu(val)) and masks h in place.

__global__ __launch_bounds__(256) void topk_kernel(
    float* __restrict__ h,            // in: pre_acts, out: masked relu'd h
    int* __restrict__ tk_idx, float* __restrict__ tk_val) {
  __shared__ int hist[256];
  __shared__ int cnts[256];
  __shared__ unsigned sh_prefix;
  __shared__ int sh_remain;
  __shared__ int sh_tie;

  const int tid = threadIdx.x;
  const int row = blockIdx.x;
  float* p = h + (size_t)row * D_SAE;

  unsigned prefix = 0;
  int remain = TOPK;

#pragma unroll
  for (int pass = 0; pass < 4; ++pass) {
    const int shift = 24 - 8 * pass;
    hist[tid] = 0;
    __syncthreads();
    for (int j = tid; j < D_SAE; j += 256) {
      unsigned k = fkey(p[j]);
      bool match = (pass == 0) || ((k >> (shift + 8)) == (prefix >> (shift + 8)));
      if (match) atomicAdd(&hist[(k >> shift) & 255], 1);
    }
    __syncthreads();
    if (tid == 0) {
      int t = 0;
      for (int b = 255; b >= 0; --b) {
        int nt = t + hist[b];
        if (nt >= remain) {
          sh_prefix = prefix | ((unsigned)b << shift);
          sh_remain = remain - t;
          break;
        }
        t = nt;
      }
      sh_tie = 0;
    }
    __syncthreads();
    prefix = sh_prefix;
    remain = sh_remain;
    __syncthreads();
  }

  const unsigned T = prefix;  // key of the K-th largest; `remain` ties needed

  // pass A: decide selection per element in this thread's contiguous chunk
  const int jbase = tid * (D_SAE / 256);
  unsigned long long m0 = 0, m1 = 0;
  int cnt = 0;
  for (int i = 0; i < D_SAE / 256; ++i) {
    unsigned k = fkey(p[jbase + i]);
    int sel = 0;
    if (k > T) sel = 1;
    else if (k == T) sel = (atomicAdd(&sh_tie, 1) < remain) ? 1 : 0;
    if (sel) {
      if (i < 64) m0 |= (1ull << i); else m1 |= (1ull << (i - 64));
      ++cnt;
    }
  }
  cnts[tid] = cnt;
  __syncthreads();
  if (tid == 0) {  // exclusive scan -> deterministic output order by index
    int run = 0;
    for (int i = 0; i < 256; ++i) { int c = cnts[i]; cnts[i] = run; run += c; }
  }
  __syncthreads();

  int off = cnts[tid];
  for (int i = 0; i < D_SAE / 256; ++i) {
    int j = jbase + i;
    bool sel = (i < 64) ? ((m0 >> i) & 1ull) : ((m1 >> (i - 64)) & 1ull);
    if (sel) {
      float v = fmaxf(p[j], 0.0f);
      p[j] = v;
      int slot = row * TOPK + off;
      tk_idx[slot] = j;
      tk_val[slot] = v;
      ++off;
    } else {
      p[j] = 0.0f;
    }
  }
}

// --------------------------- decode + per-row error ------------------------
// x_hat[n,:] = b_dec + sum_k val[k] * WdT[idx[k], :]   (coalesced row gathers)

__global__ __launch_bounds__(256) void decode_kernel(
    const int* __restrict__ tk_idx, const float* __restrict__ tk_val,
    const float* __restrict__ WdT, const float* __restrict__ b_dec,
    const float* __restrict__ x, float* __restrict__ x_hat,
    float* __restrict__ row_err) {
  __shared__ int   s_idx[TOPK];
  __shared__ float s_val[TOPK];
  __shared__ float s_part[8];

  const int tid = threadIdx.x;
  const int row = blockIdx.x;
  if (tid < TOPK) {
    s_idx[tid] = tk_idx[row * TOPK + tid];
    s_val[tid] = tk_val[row * TOPK + tid];
  }
  __syncthreads();

  float acc[D_IN / 256];
#pragma unroll
  for (int u = 0; u < D_IN / 256; ++u) acc[u] = b_dec[tid + u * 256];

  for (int k = 0; k < TOPK; ++k) {
    const float v = s_val[k];
    const float* wrow = WdT + (size_t)s_idx[k] * D_IN;
#pragma unroll
    for (int u = 0; u < D_IN / 256; ++u) acc[u] += v * wrow[tid + u * 256];
  }

  float err = 0.0f;
#pragma unroll
  for (int u = 0; u < D_IN / 256; ++u) {
    int d = tid + u * 256;
    float xh = acc[u];
    x_hat[(size_t)row * D_IN + d] = xh;
    float df = xh - x[(size_t)row * D_IN + d];
    err += df * df;
  }
  // wave32 reduce, then cross-wave via LDS
  for (int o = 16; o > 0; o >>= 1) err += __shfl_down(err, o, 32);
  if ((tid & 31) == 0) s_part[tid >> 5] = err;
  __syncthreads();
  if (tid == 0) {
    float t = 0.0f;
#pragma unroll
    for (int w = 0; w < 8; ++w) t += s_part[w];
    row_err[row] = t;
  }
}

__global__ __launch_bounds__(256) void loss_kernel(
    const float* __restrict__ row_err, float* __restrict__ loss) {
  __shared__ float s_part[8];
  const int tid = threadIdx.x;
  float t = 0.0f;
  for (int j = tid; j < N_ROWS; j += 256) t += row_err[j];
  for (int o = 16; o > 0; o >>= 1) t += __shfl_down(t, o, 32);
  if ((tid & 31) == 0) s_part[tid >> 5] = t;
  __syncthreads();
  if (tid == 0) {
    float s = 0.0f;
#pragma unroll
    for (int w = 0; w < 8; ++w) s += s_part[w];
    loss[0] = s * (1.0f / ((float)N_ROWS * (float)D_IN));
  }
}

// --------------------------- launch ----------------------------------------

extern "C" void kernel_launch(void* const* d_in, const int* in_sizes, int n_in,
                              void* d_out, int out_size, void* d_ws, size_t ws_size,
                              hipStream_t stream) {
  const float* x     = (const float*)d_in[0];
  const float* W_enc = (const float*)d_in[1];
  const float* b_enc = (const float*)d_in[2];
  const float* W_dec = (const float*)d_in[3];
  const float* b_dec = (const float*)d_in[4];

  float* x_hat = (float*)d_out;
  float* h     = x_hat + (size_t)N_ROWS * D_IN;
  float* loss  = h + (size_t)N_ROWS * D_SAE;

  // workspace layout (bytes)
  char* ws = (char*)d_ws;
  unsigned short* Xc  = (unsigned short*)ws;                                   // 16 MiB
  unsigned short* We  = (unsigned short*)(ws + (size_t)N_ROWS * D_IN * 2);     // 128 MiB
  float* WdT = (float*)(ws + (size_t)N_ROWS * D_IN * 2 + (size_t)D_SAE * D_IN * 2);
  char*  p2  = (char*)WdT + (size_t)D_SAE * D_IN * 4;
  int*   tk_idx = (int*)p2;
  float* tk_val = (float*)(p2 + (size_t)N_ROWS * TOPK * 4);
  float* row_err = (float*)(p2 + (size_t)N_ROWS * TOPK * 8);

  prep_x_kernel<<<4096, 256, 0, stream>>>(x, b_dec, Xc);
  prep_we_kernel<<<16384, 256, 0, stream>>>(W_enc, We);
  transpose_wdec_kernel<<<dim3(D_IN / 32, D_SAE / 32), dim3(32, 8), 0, stream>>>(W_dec, WdT);
  encode_gemm_kernel<<<dim3(D_SAE / BN, N_ROWS / BM), 256, 0, stream>>>(Xc, We, b_enc, h);
  topk_kernel<<<N_ROWS, 256, 0, stream>>>(h, tk_idx, tk_val);
  decode_kernel<<<N_ROWS, 256, 0, stream>>>(tk_idx, tk_val, WdT, b_dec, x, x_hat, row_err);
  loss_kernel<<<1, 256, 0, stream>>>(row_err, loss);
}
